// SelfAttentionLayer_21973052686568
// MI455X (gfx1250) — matrix operand order repaired and checked by reading, not compiled, not run on hardware
//
#include <hip/hip_runtime.h>
#include <math.h>

// MI455X / gfx1250, wave32. Fused QKV-GEMM (WMMA f16->f32) + per-node
// 4-head mixing attention + residual + LayerNorm.
//
//  * xs/qkv LDS rows padded to stride 66 floats -> conflict-free wave access
//    (bank = (2*lane + c) mod 64) in fragment build, phase-2 row reads, and
//    C-tile stores.
//  * Weights staged in WMMA *fragment-major* f16 layout ([frag][part][lane],
//    16B/lane): inner GEMM loop is pure consecutive-lane ds_load_b128 with no
//    address math and no bank conflicts; no padding needed (24.5KB).
//  * Total LDS = 161,024 B <= 160KB -> 2 workgroups/WGP residency.
//  * x tile staged via global_load_async_to_lds_b64 (ASYNCcnt path); the
//    builtin's pointer params are v2i in AS1/AS3 per the round-2 diagnostic.

typedef __attribute__((ext_vector_type(16))) _Float16 v16h;
typedef __attribute__((ext_vector_type(8)))  _Float16 v8h;
typedef __attribute__((ext_vector_type(8)))  float    v8f;
typedef __attribute__((ext_vector_type(2)))  int      v2i;

#define NODES_PER_BLOCK 128
#define THREADS         256
#define CH              64          // C_IN == C_OUT
#define LN_EPS          1e-5f
#define XS_STRIDE       66          // padded row stride in floats (bank spread)
#define XS_FLOATS       (NODES_PER_BLOCK * XS_STRIDE)   // 8448
#define QKV_PLANE       XS_FLOATS

// LDS layout (bytes):
//   Wfrag : 3*4*2 frags * 32 lanes * 32B  = 24576   (f16, fragment-major)
//   bias  : 3*64 f32                      =   768
//   gamma : 64 f32                        =   256
//   beta  : 64 f32                        =   256
//   xs    : 128*66 f32                    = 33792
//   qkv   : 3*128*66 f32                  = 101376
//   total                                 = 161024  (<= 160KB -> 2 WGs/WGP)
#define SMEM_BYTES (24576 + (3*64 + 64 + 64 + XS_FLOATS + 3*QKV_PLANE) * 4)

#if defined(__gfx1250__) && __has_builtin(__builtin_amdgcn_global_load_async_to_lds_b64) && __has_builtin(__builtin_amdgcn_s_wait_asynccnt)
#define HAVE_ASYNC_LDS 1
#else
#define HAVE_ASYNC_LDS 0
#endif

__global__ __launch_bounds__(THREADS, 1)
void fused_qkv_attn_ln(const float* __restrict__ x,
                       const float* __restrict__ Wq, const float* __restrict__ bq,
                       const float* __restrict__ Wk, const float* __restrict__ bk,
                       const float* __restrict__ Wv, const float* __restrict__ bv,
                       const float* __restrict__ gamma, const float* __restrict__ beta,
                       float* __restrict__ out, int n)
{
    extern __shared__ char smem[];
    v8h*   Wf8   = (v8h*)smem;                     // 1536 chunks of 16B
    float* biasL = (float*)(smem + 24576);         // [3][64]
    float* gammaL = biasL + 3*64;                  // [64]
    float* betaL  = gammaL + 64;                   // [64]
    float* xs     = betaL + 64;                    // [128][66]
    float* qkvL   = xs + XS_FLOATS;                // [3][128][66]

    const int tid  = threadIdx.x;
    const long base = (long)blockIdx.x * NODES_PER_BLOCK;
    const int nNodes = (int)((n - base) < NODES_PER_BLOCK ? (n - base) : NODES_PER_BLOCK);

    // ---- Phase 0a: weights -> f16 fragment-major LDS ----
    // Fragment (w matrix, t col-tile, s k-step): per lane 16 halves,
    //   B[k][nc] = W[o = 16t + nc][c = 32s + k],  nc = lane%16, k = (lane/16)*8 + 0..7, then 16 + (lane/16)*8 + 0..7
    // Chunk id: ((((w*4 + t)*2 + s)*2 + part)*32 + lane), part selects k-halves 0..7 / 8..15.
    for (int c = tid; c < 1536; c += THREADS) {
        const int L    = c & 31;
        const int part = (c >> 5) & 1;
        const int s    = (c >> 6) & 1;
        const int t    = (c >> 7) & 3;
        const int w    = c >> 9;
        const float* Wsrc = (w == 0) ? Wq : (w == 1) ? Wk : Wv;
        const int o     = t*16 + (L & 15);
        const int cbase = s*32 + part*16 + (L >> 4)*8;
        const float4* src = (const float4*)(Wsrc + o*CH + cbase);  // 8 consecutive f32
        const float4 f0 = src[0], f1 = src[1];
        v8h h;
        h[0] = (_Float16)f0.x; h[1] = (_Float16)f0.y;
        h[2] = (_Float16)f0.z; h[3] = (_Float16)f0.w;
        h[4] = (_Float16)f1.x; h[5] = (_Float16)f1.y;
        h[6] = (_Float16)f1.z; h[7] = (_Float16)f1.w;
        Wf8[c] = h;
    }
    if (tid < 64) {
        biasL[0*64 + tid] = bq[tid];
        biasL[1*64 + tid] = bk[tid];
        biasL[2*64 + tid] = bv[tid];
        gammaL[tid] = gamma[tid];
        betaL[tid]  = beta[tid];
    }

    // ---- Phase 0b: x tile -> LDS (padded rows), async-to-LDS ----
    {
        const float2* xg = (const float2*)(x + base * CH);
        const int nv2 = nNodes * (CH/2);               // float2 elements
        for (int e = tid; e < nv2; e += THREADS) {
            const int row = e >> 5, col2 = e & 31;
            float* dst = xs + row*XS_STRIDE + col2*2;
#if HAVE_ASYNC_LDS
            __builtin_amdgcn_global_load_async_to_lds_b64(
                (__attribute__((address_space(1))) v2i*)(xg + e),
                (__attribute__((address_space(3))) v2i*)dst, 0, 0);
#else
            *(float2*)dst = xg[e];
#endif
        }
    }
#if HAVE_ASYNC_LDS
    __builtin_amdgcn_s_wait_asynccnt(0);
#endif
    __syncthreads();

    // ---- Phase 1: Q/K/V GEMMs via v_wmma_f32_16x16x32_f16 ----
    {
        const int wave = tid >> 5;       // nodes [16*wave, 16*wave+16)
        const int lane = tid & 31;
        const int m    = lane & 15;      // A row / B col / C,D col
        const int grp  = lane >> 4;

        // A fragments (ISA 16-bit A 16x32 layout), built from padded xs rows.
        v16h a0, a1;
        const float* xrow = xs + (wave*16 + m) * XS_STRIDE;
        #pragma unroll
        for (int v = 0; v < 8; ++v) {
            const int kk = ((v >> 2) << 4) + (grp << 3) + ((v & 3) << 1);
            a0[2*v+0] = (_Float16)xrow[kk + 0];
            a0[2*v+1] = (_Float16)xrow[kk + 1];
            a1[2*v+0] = (_Float16)xrow[32 + kk + 0];
            a1[2*v+1] = (_Float16)xrow[32 + kk + 1];
        }

        union BV { v16h v; v8h p[2]; };

        #pragma unroll
        for (int w = 0; w < 3; ++w) {
            float* dst = qkvL + w * QKV_PLANE;
            #pragma unroll
            for (int t = 0; t < 4; ++t) {
                const int fbase = ((w*4 + t)*2) * 64;      // chunk index of (w,t,s=0)
                BV b0, b1;
                b0.p[0] = Wf8[fbase +       lane];
                b0.p[1] = Wf8[fbase +  32 + lane];
                b1.p[0] = Wf8[fbase +  64 + lane];
                b1.p[1] = Wf8[fbase +  96 + lane];

                const float bval = biasL[w*64 + t*16 + m];
                v8f acc;
                #pragma unroll
                for (int j = 0; j < 8; ++j) acc[j] = bval;  // bias pre-loaded in C
                acc = __builtin_amdgcn_wmma_f32_16x16x32_f16(
                          false, a0, false, b0.v, (short)0, acc, false, false);
                acc = __builtin_amdgcn_wmma_f32_16x16x32_f16(
                          false, a1, false, b1.v, (short)0, acc, false, false);
                // C/D layout: VGPR j, lanes 0-15 -> M=j, lanes 16-31 -> M=j+8; N=lane%16
                #pragma unroll
                for (int j = 0; j < 8; ++j) {
                    const int mm = j + (grp << 3);
                    dst[(wave*16 + mm)*XS_STRIDE + t*16 + m] = acc[j];
                }
            }
        }
    }
    __syncthreads();

    // ---- Phase 2: per-node head-mixing attention + residual + LayerNorm ----
    if (tid < nNodes) {
        const float* qn = qkvL + 0*QKV_PLANE + tid*XS_STRIDE;
        const float* kn = qkvL + 1*QKV_PLANE + tid*XS_STRIDE;
        const float* vn = qkvL + 2*QKV_PLANE + tid*XS_STRIDE;
        float* yrow = xs + tid*XS_STRIDE;

        float p[16];
        #pragma unroll
        for (int h = 0; h < 4; ++h) {
            float s[4];
            #pragma unroll
            for (int g = 0; g < 4; ++g) {
                float acc = 0.f;
                #pragma unroll
                for (int d = 0; d < 16; ++d)
                    acc += qn[h*16 + d] * kn[g*16 + d];
                s[g] = acc * 0.25f;                  // 1/sqrt(HEAD_DIM=16)
            }
            const float mx = fmaxf(fmaxf(s[0], s[1]), fmaxf(s[2], s[3]));
            float sum = 0.f;
            #pragma unroll
            for (int g = 0; g < 4; ++g) { s[g] = __expf(s[g] - mx); sum += s[g]; }
            const float inv = 1.f / sum;
            #pragma unroll
            for (int g = 0; g < 4; ++g) p[h*4 + g] = s[g] * inv;
        }

        float mu = 0.f, m2 = 0.f;
        #pragma unroll
        for (int c = 0; c < CH; ++c) {
            const int h = c >> 4, d = c & 15;
            const float a = p[h*4+0]*vn[0*16+d] + p[h*4+1]*vn[1*16+d]
                          + p[h*4+2]*vn[2*16+d] + p[h*4+3]*vn[3*16+d];
            const float yv = a + yrow[c];            // residual (xs holds x)
            yrow[c] = yv;
            mu += yv; m2 += yv * yv;
        }
        mu *= (1.f/64.f);
        const float var  = m2*(1.f/64.f) - mu*mu;
        const float rstd = rsqrtf(var + LN_EPS);
        #pragma unroll
        for (int c = 0; c < CH; ++c)
            yrow[c] = (yrow[c] - mu) * rstd * gammaL[c] + betaL[c];
    }
    __syncthreads();

    // ---- Phase 3: coalesced b128 writeback from padded rows ----
    {
        float4* og = (float4*)(out + base * CH);
        const int nvec = nNodes * (CH/4);
        for (int g = tid; g < nvec; g += THREADS) {
            const int row = g >> 4, col = (g & 15) * 4;
            const float* src = xs + row*XS_STRIDE + col;   // 8B-aligned
            const float2 u0 = *(const float2*)(src);
            const float2 u1 = *(const float2*)(src + 2);
            float4 o4; o4.x = u0.x; o4.y = u0.y; o4.z = u1.x; o4.w = u1.y;
            og[g] = o4;
        }
    }
}

extern "C" void kernel_launch(void* const* d_in, const int* in_sizes, int n_in,
                              void* d_out, int out_size, void* d_ws, size_t ws_size,
                              hipStream_t stream) {
    (void)n_in; (void)out_size; (void)d_ws; (void)ws_size;
    const float* x     = (const float*)d_in[0];
    /* d_in[1] = batch (int64) is unused by the reference math */
    const float* Wq    = (const float*)d_in[2];
    const float* bq    = (const float*)d_in[3];
    const float* Wk    = (const float*)d_in[4];
    const float* bk    = (const float*)d_in[5];
    const float* Wv    = (const float*)d_in[6];
    const float* bv    = (const float*)d_in[7];
    const float* gamma = (const float*)d_in[8];
    const float* beta  = (const float*)d_in[9];
    float* out = (float*)d_out;

    const int n = in_sizes[0] / CH;          // number of nodes
    const int grid = (n + NODES_PER_BLOCK - 1) / NODES_PER_BLOCK;

    (void)hipFuncSetAttribute(reinterpret_cast<const void*>(fused_qkv_attn_ln),
                              hipFuncAttributeMaxDynamicSharedMemorySize, SMEM_BYTES);
    fused_qkv_attn_ln<<<grid, THREADS, SMEM_BYTES, stream>>>(
        x, Wq, bq, Wk, bk, Wv, bv, gamma, beta, out, n);
}
